// MultiHeadAttention_40381282517699
// MI455X (gfx1250) — compile-verified, matrix operand order checked by reading, not run on hardware
//
#include <hip/hip_runtime.h>

// Problem constants
#define B_DIM   2
#define S_DIM   2048
#define E_DIM   1024
#define H_DIM   16
#define HD_DIM  64
#define M_TOT   (B_DIM * S_DIM)      // 4096
#define N_QKV   (3 * E_DIM)          // 3072

typedef __attribute__((ext_vector_type(16))) __bf16      bf16x16;
typedef __attribute__((ext_vector_type(8)))  float        f32x8;
typedef __attribute__((ext_vector_type(4)))  unsigned int u32x4;
typedef int i32x4 __attribute__((vector_size(16)));

// ---- CDNA5 async global->LDS path (ASYNCcnt), with safe fallback ----
#if defined(__has_builtin)
#  if __has_builtin(__builtin_amdgcn_global_load_async_to_lds_b128) && \
      __has_builtin(__builtin_amdgcn_s_wait_asynccnt)
#    define USE_ASYNC_LDS 1
#  endif
#endif
#ifndef USE_ASYNC_LDS
#  define USE_ASYNC_LDS 0
#endif

#if USE_ASYNC_LDS
typedef __attribute__((address_space(1))) i32x4 gvec4i;
typedef __attribute__((address_space(3))) i32x4 lvec4i;
#endif

__device__ __forceinline__ void copy16_g2l(const __bf16* g, __bf16* l) {
#if USE_ASYNC_LDS
  __builtin_amdgcn_global_load_async_to_lds_b128(
      (gvec4i*)(void*)g, (lvec4i*)(void*)l, 0, 0);
#else
  *(u32x4*)l = *(const u32x4*)g;
#endif
}
__device__ __forceinline__ void wait_async0() {
#if USE_ASYNC_LDS
  __builtin_amdgcn_s_wait_asynccnt(0);
#endif
}

__device__ __forceinline__ f32x8 zero8() {
  f32x8 z = {0.f, 0.f, 0.f, 0.f, 0.f, 0.f, 0.f, 0.f};
  return z;
}

__device__ __forceinline__ f32x8 wmma_bf16(bf16x16 a, bf16x16 b, f32x8 c) {
  return __builtin_amdgcn_wmma_f32_16x16x32_bf16(
      false, a, false, b, (short)0, c, false, false);
}

// 16x32 bf16 operand fragment in the documented A/B VGPR layout.
__device__ __forceinline__ bf16x16 load_frag(const __bf16* __restrict__ base,
                                             int row0, int ld, int k0, int lane) {
  int r  = lane & 15;
  int hi = (lane >> 4) & 1;
  const __bf16* p = base + (size_t)(row0 + r) * (size_t)ld + (size_t)(k0 + (hi << 3));
  union { bf16x16 v; u32x4 u[2]; } f;
  f.u[0] = *(const u32x4*)(p);
  f.u[1] = *(const u32x4*)(p + 16);
  return f.v;
}

// half-wave reductions (lanes 0-15 / 16-31 hold distinct C rows)
__device__ __forceinline__ float half16_max(float v) {
  v = fmaxf(v, __shfl_xor(v, 1, 32));
  v = fmaxf(v, __shfl_xor(v, 2, 32));
  v = fmaxf(v, __shfl_xor(v, 4, 32));
  v = fmaxf(v, __shfl_xor(v, 8, 32));
  return v;
}
__device__ __forceinline__ float half16_sum(float v) {
  v += __shfl_xor(v, 1, 32);
  v += __shfl_xor(v, 2, 32);
  v += __shfl_xor(v, 4, 32);
  v += __shfl_xor(v, 8, 32);
  return v;
}

// ---------------- Kernel A: f32 -> bf16 conversion ----------------
__global__ void mha_cvt_bf16(const float* __restrict__ src, __bf16* __restrict__ dst, int n) {
  int i = (blockIdx.x * blockDim.x + threadIdx.x) * 4;
  if (i + 3 < n) {
    float4 v = *(const float4*)(src + i);
    dst[i + 0] = (__bf16)v.x;
    dst[i + 1] = (__bf16)v.y;
    dst[i + 2] = (__bf16)v.z;
    dst[i + 3] = (__bf16)v.w;
  }
}

// ---------------- Kernel B: QKV projection GEMM ----------------
// y[m][n] = sum_k x[m][k] * w_qkv[n][k]; M=4096, N=3072, K=1024.
// Workgroup 256(M) x 64(N); each wave 32x64 (2 A-frags x 4 B-frags = 8 WMMA).
// B panel (64x32 per k-step) staged in LDS via async loads, double-buffered.
__global__ void mha_qkv_gemm(const __bf16* __restrict__ xb,
                             const __bf16* __restrict__ wb,
                             __bf16* __restrict__ qb,
                             __bf16* __restrict__ kb,
                             __bf16* __restrict__ vtb) {
  __shared__ __bf16 bpanel[2][64 * 32];

  int tid  = threadIdx.x;
  int lane = tid & 31;
  int wave = tid >> 5;
  int m0 = blockIdx.y * 256 + wave * 32;
  int n0 = blockIdx.x * 64;
  int hi = (lane >> 4) & 1;

  int srow = tid >> 2;        // 0..63
  int sch  = (tid & 3) << 3;  // 0/8/16/24

  f32x8 acc[2][4];
#pragma unroll
  for (int i = 0; i < 2; ++i)
#pragma unroll
    for (int t = 0; t < 4; ++t) acc[i][t] = zero8();

  const int KSTEPS = E_DIM / 32;
  copy16_g2l(wb + (size_t)(n0 + srow) * E_DIM + sch, &bpanel[0][srow * 32 + sch]);

#pragma unroll 1
  for (int kk = 0; kk < KSTEPS; ++kk) {
    wait_async0();
    __syncthreads();
    if (kk + 1 < KSTEPS)
      copy16_g2l(wb + (size_t)(n0 + srow) * E_DIM + (kk + 1) * 32 + sch,
                 &bpanel[(kk + 1) & 1][srow * 32 + sch]);

    const __bf16* bp = &bpanel[kk & 1][0];
    // prefetch all fragments, then WMMA burst
    bf16x16 a0 = load_frag(xb, m0,      E_DIM, kk * 32, lane);
    bf16x16 a1 = load_frag(xb, m0 + 16, E_DIM, kk * 32, lane);
    bf16x16 b0 = load_frag(bp,  0, 32, 0, lane);
    bf16x16 b1 = load_frag(bp, 16, 32, 0, lane);
    bf16x16 b2 = load_frag(bp, 32, 32, 0, lane);
    bf16x16 b3 = load_frag(bp, 48, 32, 0, lane);
    acc[0][0] = wmma_bf16(a0, b0, acc[0][0]);
    acc[1][0] = wmma_bf16(a1, b0, acc[1][0]);
    acc[0][1] = wmma_bf16(a0, b1, acc[0][1]);
    acc[1][1] = wmma_bf16(a1, b1, acc[1][1]);
    acc[0][2] = wmma_bf16(a0, b2, acc[0][2]);
    acc[1][2] = wmma_bf16(a1, b2, acc[1][2]);
    acc[0][3] = wmma_bf16(a0, b3, acc[0][3]);
    acc[1][3] = wmma_bf16(a1, b3, acc[1][3]);
  }

#pragma unroll
  for (int i = 0; i < 2; ++i) {
#pragma unroll
    for (int t = 0; t < 4; ++t) {
#pragma unroll
      for (int r = 0; r < 8; ++r) {
        int col = n0 + t * 16 + (lane & 15);
        int h = col / 192;
        int rem = col - h * 192;
        int which = rem >> 6;     // 0=Q, 1=K, 2=V
        int d = rem & 63;
        int m = m0 + i * 16 + r + (hi << 3);
        int bi = m >> 11;
        int s = m & (S_DIM - 1);
        __bf16 v = (__bf16)acc[i][t][r];
        if (which == 0)
          qb[(((size_t)bi * H_DIM + h) * S_DIM + s) * HD_DIM + d] = v;
        else if (which == 1)
          kb[(((size_t)bi * H_DIM + h) * S_DIM + s) * HD_DIM + d] = v;
        else
          vtb[(((size_t)bi * H_DIM + h) * HD_DIM + d) * S_DIM + s] = v;
      }
    }
  }
}

// ---------------- Kernel C: causal flash attention ----------------
// grid: (S/128, B*H); block 256 = 8 waves; each wave owns 16 query rows.
// KV blocks of 64 rows staged in LDS (async, double-buffered), shared by all
// 8 waves; staging loop workgroup-uniform, compute skipped per-wave beyond
// the causal frontier.
__global__ void mha_flash_attn(const __bf16* __restrict__ qb,
                               const __bf16* __restrict__ kb,
                               const __bf16* __restrict__ vtb,
                               __bf16* __restrict__ ob) {
  __shared__ __bf16 kpanel[2][64 * 64];   // K rows kv0..+63, 64 dims
  __shared__ __bf16 vpanel[2][64 * 64];   // Vt rows d=0..63, kv cols kv0..+63
  __shared__ __bf16 p_lds[8][16 * 64];    // per-wave P staging (16x64)

  int tid  = threadIdx.x;
  int lane = tid & 31;
  int wave = tid >> 5;
  int hi = (lane >> 4) & 1;

  int bh = blockIdx.y;
  int bi = bh / H_DIM;
  int h  = bh % H_DIM;
  const __bf16* qba = qb  + (size_t)bh * S_DIM * HD_DIM;
  const __bf16* kba = kb  + (size_t)bh * S_DIM * HD_DIM;
  const __bf16* vba = vtb + (size_t)bh * HD_DIM * S_DIM;

  int qblk0 = blockIdx.x * 128;
  int q0 = qblk0 + wave * 16;

  // staging: 512 16B chunks per panel, 2 per thread
  int row0c = tid >> 3;              // 0..31
  int ch0   = (tid & 7) << 3;        // 0..56

  bf16x16 aq0 = load_frag(qba, q0, HD_DIM, 0, lane);
  bf16x16 aq1 = load_frag(qba, q0, HD_DIM, 32, lane);

  f32x8 o_acc[4];
#pragma unroll
  for (int t = 0; t < 4; ++t) o_acc[t] = zero8();
  float m_r[8], l_r[8];
#pragma unroll
  for (int r = 0; r < 8; ++r) { m_r[r] = -__builtin_inff(); l_r[r] = 0.f; }

  const float scale = 0.125f;
  __bf16* pl = &p_lds[wave][0];
  int nblk = (qblk0 + 128) >> 6;     // uniform across workgroup (2 or more)

  // prologue: stage kv block 0 (K rows 0..63, Vt cols 0..63)
  copy16_g2l(kba + (size_t)row0c * HD_DIM + ch0,        &kpanel[0][row0c * 64 + ch0]);
  copy16_g2l(kba + (size_t)(row0c + 32) * HD_DIM + ch0, &kpanel[0][(row0c + 32) * 64 + ch0]);
  copy16_g2l(vba + (size_t)row0c * S_DIM + ch0,         &vpanel[0][row0c * 64 + ch0]);
  copy16_g2l(vba + (size_t)(row0c + 32) * S_DIM + ch0,  &vpanel[0][(row0c + 32) * 64 + ch0]);

#pragma unroll 1
  for (int blk = 0; blk < nblk; ++blk) {
    wait_async0();
    __syncthreads();
    if (blk + 1 < nblk) {
      int nkv = (blk + 1) << 6;
      int nb = (blk + 1) & 1;
      copy16_g2l(kba + (size_t)(nkv + row0c) * HD_DIM + ch0,
                 &kpanel[nb][row0c * 64 + ch0]);
      copy16_g2l(kba + (size_t)(nkv + row0c + 32) * HD_DIM + ch0,
                 &kpanel[nb][(row0c + 32) * 64 + ch0]);
      copy16_g2l(vba + (size_t)row0c * S_DIM + nkv + ch0,
                 &vpanel[nb][row0c * 64 + ch0]);
      copy16_g2l(vba + (size_t)(row0c + 32) * S_DIM + nkv + ch0,
                 &vpanel[nb][(row0c + 32) * 64 + ch0]);
    }

    int kv0 = blk << 6;
    if (kv0 < q0 + 16) {            // per-wave causal skip (no barriers inside)
      const __bf16* kp = &kpanel[blk & 1][0];
      const __bf16* vp = &vpanel[blk & 1][0];

      // ---- scores: 4 sub-tiles of 16 kv cols, K-dim 64 = 2 WMMA each
      f32x8 sv[4];
#pragma unroll
      for (int nt = 0; nt < 4; ++nt) {
        bf16x16 bk0 = load_frag(kp, nt * 16, 64, 0, lane);
        bf16x16 bk1 = load_frag(kp, nt * 16, 64, 32, lane);
        f32x8 sc = zero8();
        sc = wmma_bf16(aq0, bk0, sc);
        sc = wmma_bf16(aq1, bk1, sc);
#pragma unroll
        for (int r = 0; r < 8; ++r) {
          float v = sc[r] * scale;
          int row = q0 + r + (hi << 3);
          int col = kv0 + nt * 16 + (lane & 15);
          if (col > row) v = -__builtin_inff();
          sc[r] = v;
        }
        sv[nt] = sc;
      }

      // ---- online softmax
      float alpha[8];
#pragma unroll
      for (int r = 0; r < 8; ++r) {
        float mx = fmaxf(fmaxf(sv[0][r], sv[1][r]), fmaxf(sv[2][r], sv[3][r]));
        mx = half16_max(mx);
        float mnew = fmaxf(m_r[r], mx);
        alpha[r] = __expf(m_r[r] - mnew);
        m_r[r] = mnew;
      }
#pragma unroll
      for (int r = 0; r < 8; ++r) {
        float rs = 0.f;
#pragma unroll
        for (int nt = 0; nt < 4; ++nt) {
          float p = __expf(sv[nt][r] - m_r[r]);
          sv[nt][r] = p;
          rs += p;
        }
        rs = half16_sum(rs);
        l_r[r] = l_r[r] * alpha[r] + rs;
#pragma unroll
        for (int t = 0; t < 4; ++t) o_acc[t][r] *= alpha[r];
      }

      // ---- P: C-layout -> LDS row-major (16x64) -> A-layout
#pragma unroll
      for (int nt = 0; nt < 4; ++nt)
#pragma unroll
        for (int r = 0; r < 8; ++r)
          pl[(r + (hi << 3)) * 64 + nt * 16 + (lane & 15)] = (__bf16)sv[nt][r];
      __builtin_amdgcn_wave_barrier();
      bf16x16 ap0 = load_frag(pl, 0, 64, 0, lane);
      bf16x16 ap1 = load_frag(pl, 0, 64, 32, lane);
      __builtin_amdgcn_wave_barrier();

      // ---- O += P @ V (K-dim 64 = 2 WMMA per output tile)
#pragma unroll
      for (int t = 0; t < 4; ++t) {
        bf16x16 bv0 = load_frag(vp, t * 16, 64, 0, lane);
        bf16x16 bv1 = load_frag(vp, t * 16, 64, 32, lane);
        o_acc[t] = wmma_bf16(ap0, bv0, o_acc[t]);
        o_acc[t] = wmma_bf16(ap1, bv1, o_acc[t]);
      }
    }
  }

  // ---- normalize and store (B,S,E) bf16
#pragma unroll
  for (int t = 0; t < 4; ++t) {
#pragma unroll
    for (int r = 0; r < 8; ++r) {
      int row = q0 + r + (hi << 3);
      int e = h * HD_DIM + t * 16 + (lane & 15);
      ob[((size_t)bi * S_DIM + row) * E_DIM + e] = (__bf16)(o_acc[t][r] / l_r[r]);
    }
  }
}

// ---------------- Kernel D: output projection + bias ----------------
__global__ void mha_proj_gemm(const __bf16* __restrict__ ob,
                              const __bf16* __restrict__ wpb,
                              const float* __restrict__ bias,
                              float* __restrict__ out) {
  __shared__ __bf16 bpanel[2][64 * 32];

  int tid  = threadIdx.x;
  int lane = tid & 31;
  int wave = tid >> 5;
  int m0 = blockIdx.y * 256 + wave * 32;
  int n0 = blockIdx.x * 64;
  int hi = (lane >> 4) & 1;

  int srow = tid >> 2;
  int sch  = (tid & 3) << 3;

  f32x8 acc[2][4];
#pragma unroll
  for (int i = 0; i < 2; ++i)
#pragma unroll
    for (int t = 0; t < 4; ++t) acc[i][t] = zero8();

  const int KSTEPS = E_DIM / 32;
  copy16_g2l(wpb + (size_t)(n0 + srow) * E_DIM + sch, &bpanel[0][srow * 32 + sch]);

#pragma unroll 1
  for (int kk = 0; kk < KSTEPS; ++kk) {
    wait_async0();
    __syncthreads();
    if (kk + 1 < KSTEPS)
      copy16_g2l(wpb + (size_t)(n0 + srow) * E_DIM + (kk + 1) * 32 + sch,
                 &bpanel[(kk + 1) & 1][srow * 32 + sch]);

    const __bf16* bp = &bpanel[kk & 1][0];
    bf16x16 a0 = load_frag(ob, m0,      E_DIM, kk * 32, lane);
    bf16x16 a1 = load_frag(ob, m0 + 16, E_DIM, kk * 32, lane);
    bf16x16 b0 = load_frag(bp,  0, 32, 0, lane);
    bf16x16 b1 = load_frag(bp, 16, 32, 0, lane);
    bf16x16 b2 = load_frag(bp, 32, 32, 0, lane);
    bf16x16 b3 = load_frag(bp, 48, 32, 0, lane);
    acc[0][0] = wmma_bf16(a0, b0, acc[0][0]);
    acc[1][0] = wmma_bf16(a1, b0, acc[1][0]);
    acc[0][1] = wmma_bf16(a0, b1, acc[0][1]);
    acc[1][1] = wmma_bf16(a1, b1, acc[1][1]);
    acc[0][2] = wmma_bf16(a0, b2, acc[0][2]);
    acc[1][2] = wmma_bf16(a1, b2, acc[1][2]);
    acc[0][3] = wmma_bf16(a0, b3, acc[0][3]);
    acc[1][3] = wmma_bf16(a1, b3, acc[1][3]);
  }

#pragma unroll
  for (int i = 0; i < 2; ++i) {
#pragma unroll
    for (int t = 0; t < 4; ++t) {
#pragma unroll
      for (int r = 0; r < 8; ++r) {
        int col = n0 + t * 16 + (lane & 15);
        int m = m0 + i * 16 + r + (hi << 3);
        out[(size_t)m * E_DIM + col] = acc[i][t][r] + bias[col];
      }
    }
  }
}

extern "C" void kernel_launch(void* const* d_in, const int* in_sizes, int n_in,
                              void* d_out, int out_size, void* d_ws, size_t ws_size,
                              hipStream_t stream) {
  const float* x      = (const float*)d_in[0];
  const float* w_qkv  = (const float*)d_in[1];
  const float* w_proj = (const float*)d_in[2];
  const float* b_proj = (const float*)d_in[3];
  float* out = (float*)d_out;

  char* ws = (char*)d_ws;
  __bf16* xb    = (__bf16*)(ws + 0);
  __bf16* wqkvb = (__bf16*)(ws + (8u  << 20));
  __bf16* wpb   = (__bf16*)(ws + (14u << 20));
  __bf16* qb    = (__bf16*)(ws + (16u << 20));
  __bf16* kb    = (__bf16*)(ws + (24u << 20));
  __bf16* vtb   = (__bf16*)(ws + (32u << 20));
  __bf16* ob    = (__bf16*)(ws + (40u << 20));

  const int n_x  = M_TOT * E_DIM;
  const int n_wq = N_QKV * E_DIM;
  const int n_wp = E_DIM * E_DIM;

  mha_cvt_bf16<<<n_x  / 1024, 256, 0, stream>>>(x, xb, n_x);
  mha_cvt_bf16<<<n_wq / 1024, 256, 0, stream>>>(w_qkv, wqkvb, n_wq);
  mha_cvt_bf16<<<n_wp / 1024, 256, 0, stream>>>(w_proj, wpb, n_wp);

  // QKV GEMM: N 3072/64 = 48, M 4096/256 = 16
  mha_qkv_gemm<<<dim3(48, 16), 256, 0, stream>>>(xb, wqkvb, qb, kb, vtb);
  // Flash attention: (S/128 = 16) x (B*H = 32)
  mha_flash_attn<<<dim3(16, 32), 256, 0, stream>>>(qb, kb, vtb, ob);
  // Output projection: N 1024/64 = 16, M 4096/256 = 16
  mha_proj_gemm<<<dim3(16, 16), 256, 0, stream>>>(ob, wpb, b_proj, out);
}
